// MeshGraphNetProcessor_68504728371501
// MI455X (gfx1250) — compile-verified, matrix operand order checked by reading, not run on hardware
//
#include <hip/hip_runtime.h>
#include <hip/hip_bf16.h>

// ---------------------------------------------------------------------------
// MeshGraphNet processor on gfx1250: bf16 WMMA (v_wmma_f32_16x16x32_bf16)
// for all MLP GEMMs, f32 accumulate / residual / scatter-add.
// Edge kernel: 64 edges/block (4 M-tiles per wave, B-fragment reuse).
// Gather phases vectorized: 8-float chunks -> one ds_store_b128 (the packed
// A-fragment layout is contiguous over aligned k-octets).
// ---------------------------------------------------------------------------

#define GP_P       15
#define GP_D       128
#define GP_NODES   10000
#define GP_EDGES   160000
#define EBM        64          // edges per block (4 M-tiles)
#define EMT        4           // edge M-tiles
#define NBM        16          // nodes per block
#define TILE_US    512         // ushorts per packed 16x32 fragment tile (32 lanes * 16)

typedef __attribute__((ext_vector_type(16))) __bf16 v16bf;
typedef __attribute__((ext_vector_type(8)))  float  v8f;

static __device__ __forceinline__ unsigned short f32_to_bf16_raw(float f) {
    unsigned int u = __float_as_uint(f);
    unsigned int lsb = (u >> 16) & 1u;
    u += 0x7fffu + lsb;                 // round-to-nearest-even
    return (unsigned short)(u >> 16);
}

static __device__ __forceinline__ unsigned int pack2_bf16(float lo, float hi) {
    return (unsigned int)f32_to_bf16_raw(lo) |
           ((unsigned int)f32_to_bf16_raw(hi) << 16);
}

// Packed offset of element (row m in 0..15, k) inside the WMMA 16-bit
// A-fragment layout. For k in an aligned octet [8q..8q+7] the offsets are
// contiguous, enabling 16-byte LDS stores.
static __device__ __forceinline__ int pack_off(int m, int k) {
    int kt = k >> 5, kk = k & 31;
    int hs = (kk >> 3) & 1;
    int j2 = ((kk >> 4) & 1) * 4 + ((kk & 7) >> 1);
    int p  = kk & 1;
    return kt * TILE_US + (hs * 16 + m) * 16 + j2 * 2 + p;
}

static __device__ __forceinline__ v8f bias8(float b) {
    v8f c;
#pragma unroll
    for (int r = 0; r < 8; ++r) c[r] = b;
    return c;
}

// Convert 8 consecutive floats to a packed bf16 octet and store 16B into LDS.
static __device__ __forceinline__ void pack8_to_lds(unsigned short* lds_dst,
                                                    const float* __restrict__ srcp) {
    float4 f0 = *(const float4*)srcp;
    float4 f1 = *(const float4*)(srcp + 4);
    uint4 u;
    u.x = pack2_bf16(f0.x, f0.y);
    u.y = pack2_bf16(f0.z, f0.w);
    u.z = pack2_bf16(f1.x, f1.y);
    u.w = pack2_bf16(f1.z, f1.w);
    *reinterpret_cast<uint4*>(lds_dst) = u;
}

// Multi-M-tile WMMA GEMM: B fragment loaded once per k-tile, reused across MT
// accumulators. A: LDS packed fragments [MT][KT][512]; B: global packed.
template <int KT, int MT>
static __device__ __forceinline__ void gemm_multi(const unsigned short* A,
                                                  const unsigned short* B,
                                                  int wn, int lane, v8f* acc) {
#pragma unroll
    for (int kt = 0; kt < KT; ++kt) {
        v16bf b = *reinterpret_cast<const v16bf*>(B + ((size_t)(kt * 8 + wn) * 32 + lane) * 16);
#pragma unroll
        for (int mt = 0; mt < MT; ++mt) {
            v16bf a = *reinterpret_cast<const v16bf*>(
                A + mt * (KT * TILE_US) + kt * TILE_US + lane * 16);
            acc[mt] = __builtin_amdgcn_wmma_f32_16x16x32_bf16(
                false, a, false, b, (short)0, acc[mt], false, false);
        }
    }
}

// ReLU + bf16 pack-store of a 16x16 C/D tile into LDS A-fragment layout.
static __device__ __forceinline__ void store_hidden(unsigned short* H,
                                                    int wn, int lane, v8f acc) {
    int hs = lane >> 4, nl = lane & 15;
    int n = wn * 16 + nl;
#pragma unroll
    for (int r = 0; r < 8; ++r) {
        float v = acc[r];
        v = v > 0.0f ? v : 0.0f;
        H[pack_off(r + 8 * hs, n)] = f32_to_bf16_raw(v);
    }
}

// f32 weights [layers][K][128] -> bf16 B-fragment-packed [layers][K/32][8][32][16]
__global__ __launch_bounds__(256) void pack_weights_kernel(
    const float* __restrict__ src, unsigned short* __restrict__ dst,
    int K, int total) {
    int idx = blockIdx.x * blockDim.x + threadIdx.x;
    if (idx >= total) return;
    int per_layer = K * GP_D;
    int layer = idx / per_layer;
    int r     = idx % per_layer;
    int j    = r & 15;
    int lane = (r >> 4) & 31;
    int nt   = (r >> 9) & 7;
    int kt   = r >> 12;
    int j2 = j >> 1, p = j & 1;
    int kk = ((j2 >= 4) ? 16 : 0) + ((lane >> 4) * 8) + ((j2 & 3) * 2) + p;
    int k  = kt * 32 + kk;
    int n  = nt * 16 + (lane & 15);
    dst[idx] = f32_to_bf16_raw(src[((size_t)layer * K + k) * GP_D + n]);
}

// ---------------------------------------------------------------------------
// Edge block: e' = MLP([n[src], n[dst], e]) + e ; scatter-add e' into agg[dst]
// 1 block = 64 edges, 8 waves; each wave owns one 16-wide N tile across 4
// M-tiles so every B fragment feeds 4 WMMAs.
// ---------------------------------------------------------------------------
__global__ __launch_bounds__(256) void edge_block_kernel(
    const float* __restrict__ nbuf, float* __restrict__ ebuf,
    float* __restrict__ agg,
    const int* __restrict__ src, const int* __restrict__ dst,
    const unsigned short* __restrict__ w0, const unsigned short* __restrict__ w1,
    const unsigned short* __restrict__ w2,
    const float* __restrict__ b0, const float* __restrict__ b1,
    const float* __restrict__ b2) {
    __shared__ unsigned short ldsA[EMT * 12 * TILE_US];   // 64 x 384 bf16, packed
    __shared__ unsigned short ldsH1[EMT * 4 * TILE_US];   // 64 x 128
    __shared__ unsigned short ldsH2[EMT * 4 * TILE_US];

    const int tid  = threadIdx.x;
    const int lane = tid & 31;
    const int wn   = tid >> 5;
    const int eb   = blockIdx.x * EBM;

    // Gather [n[src] | n[dst] | e] -> bf16 packed LDS.
    // 64*384/8 = 3072 octet-chunks, 12 per thread, one ds_store_b128 each.
#pragma unroll
    for (int i = 0; i < (EBM * 3 * GP_D) / (8 * 256); ++i) {
        int c  = tid + i * 256;
        int m  = c / 48;            // 48 octets per row of 384
        int kb = (c % 48) * 8;
        int e  = eb + m;
        const float* srcp;
        if (kb < GP_D)            srcp = nbuf + (size_t)src[e] * GP_D + kb;
        else if (kb < 2 * GP_D)   srcp = nbuf + (size_t)dst[e] * GP_D + (kb - GP_D);
        else                      srcp = ebuf + (size_t)e * GP_D + (kb - 2 * GP_D);
        int mt = m >> 4, ml = m & 15;
        pack8_to_lds(&ldsA[mt * (12 * TILE_US) + pack_off(ml, kb)], srcp);
    }
    __syncthreads();

    const int nl = lane & 15, hs = lane >> 4;
    const int n  = wn * 16 + nl;

    v8f acc[EMT];
#pragma unroll
    for (int mt = 0; mt < EMT; ++mt) acc[mt] = bias8(b0[n]);
    gemm_multi<12, EMT>(ldsA, w0, wn, lane, acc);          // 384 -> 128
#pragma unroll
    for (int mt = 0; mt < EMT; ++mt)
        store_hidden(ldsH1 + mt * (4 * TILE_US), wn, lane, acc[mt]);
    __syncthreads();

#pragma unroll
    for (int mt = 0; mt < EMT; ++mt) acc[mt] = bias8(b1[n]);
    gemm_multi<4, EMT>(ldsH1, w1, wn, lane, acc);          // 128 -> 128
#pragma unroll
    for (int mt = 0; mt < EMT; ++mt)
        store_hidden(ldsH2 + mt * (4 * TILE_US), wn, lane, acc[mt]);
    __syncthreads();

#pragma unroll
    for (int mt = 0; mt < EMT; ++mt) acc[mt] = bias8(b2[n]);
    gemm_multi<4, EMT>(ldsH2, w2, wn, lane, acc);          // 128 -> 128

    // Epilogue: residual, in-place edge update, fp32 atomic scatter into agg.
#pragma unroll
    for (int mt = 0; mt < EMT; ++mt) {
#pragma unroll
        for (int r = 0; r < 8; ++r) {
            int m = mt * 16 + r + 8 * hs;
            int e = eb + m;
            size_t off = (size_t)e * GP_D + n;
            float v = acc[mt][r] + ebuf[off];
            ebuf[off] = v;
            unsafeAtomicAdd(&agg[(size_t)dst[e] * GP_D + n], v);
        }
    }
}

// ---------------------------------------------------------------------------
// Node block: n' = MLP([n, agg]) + n   (16 nodes per block)
// ---------------------------------------------------------------------------
__global__ __launch_bounds__(256) void node_block_kernel(
    float* __restrict__ nbuf, const float* __restrict__ agg,
    const unsigned short* __restrict__ w0, const unsigned short* __restrict__ w1,
    const unsigned short* __restrict__ w2,
    const float* __restrict__ b0, const float* __restrict__ b1,
    const float* __restrict__ b2) {
    __shared__ unsigned short ldsA[8 * TILE_US];    // 16 x 256 bf16, packed
    __shared__ unsigned short ldsH1[4 * TILE_US];
    __shared__ unsigned short ldsH2[4 * TILE_US];

    const int tid  = threadIdx.x;
    const int lane = tid & 31;
    const int wn   = tid >> 5;
    const int nb   = blockIdx.x * NBM;

    // 16*256/8 = 512 octet-chunks, 2 per thread.
#pragma unroll
    for (int i = 0; i < (NBM * 2 * GP_D) / (8 * 256); ++i) {
        int c  = tid + i * 256;
        int m  = c >> 5;            // 32 octets per row of 256
        int kb = (c & 31) * 8;
        int node = nb + m;
        const float* srcp = (kb < GP_D)
            ? nbuf + (size_t)node * GP_D + kb
            : agg  + (size_t)node * GP_D + (kb - GP_D);
        pack8_to_lds(&ldsA[pack_off(m, kb)], srcp);
    }
    __syncthreads();

    const int nl = lane & 15, hs = lane >> 4;
    const int n  = wn * 16 + nl;

    v8f acc[1];
    acc[0] = bias8(b0[n]);
    gemm_multi<8, 1>(ldsA, w0, wn, lane, acc);       // 256 -> 128
    store_hidden(ldsH1, wn, lane, acc[0]);
    __syncthreads();

    acc[0] = bias8(b1[n]);
    gemm_multi<4, 1>(ldsH1, w1, wn, lane, acc);
    store_hidden(ldsH2, wn, lane, acc[0]);
    __syncthreads();

    acc[0] = bias8(b2[n]);
    gemm_multi<4, 1>(ldsH2, w2, wn, lane, acc);

#pragma unroll
    for (int r = 0; r < 8; ++r) {
        int m = r + 8 * hs;
        size_t off = (size_t)(nb + m) * GP_D + n;
        nbuf[off] = acc[0][r] + nbuf[off];
    }
}

// ---------------------------------------------------------------------------
extern "C" void kernel_launch(void* const* d_in, const int* in_sizes, int n_in,
                              void* d_out, int out_size, void* d_ws, size_t ws_size,
                              hipStream_t stream) {
    (void)in_sizes; (void)n_in; (void)out_size; (void)ws_size;

    const float* node_f = (const float*)d_in[0];
    const float* edge_f = (const float*)d_in[1];
    const int*   src    = (const int*)d_in[2];
    const int*   dst    = (const int*)d_in[3];
    const float* ew0 = (const float*)d_in[4];
    const float* eb0 = (const float*)d_in[5];
    const float* ew1 = (const float*)d_in[6];
    const float* eb1 = (const float*)d_in[7];
    const float* ew2 = (const float*)d_in[8];
    const float* eb2 = (const float*)d_in[9];
    const float* nw0 = (const float*)d_in[10];
    const float* nb0 = (const float*)d_in[11];
    const float* nw1 = (const float*)d_in[12];
    const float* nb1 = (const float*)d_in[13];
    const float* nw2 = (const float*)d_in[14];
    const float* nb2 = (const float*)d_in[15];

    // ---- workspace layout (all sizes multiples of 256 B) ----
    const size_t E_B  = (size_t)GP_EDGES * GP_D * sizeof(float);   // 81,920,000
    const size_t N_B  = (size_t)GP_NODES * GP_D * sizeof(float);   //  5,120,000
    const int EW0_E = GP_P * 3 * GP_D * GP_D;   // 737,280
    const int EW1_E = GP_P * GP_D * GP_D;       // 245,760
    const int NW0_E = GP_P * 2 * GP_D * GP_D;   // 491,520
    const int NW1_E = GP_P * GP_D * GP_D;

    char* ws = (char*)d_ws;
    float* e_buf = (float*)ws;                 ws += E_B;
    float* n_buf = (float*)ws;                 ws += N_B;
    float* agg   = (float*)ws;                 ws += N_B;
    unsigned short* ew0p = (unsigned short*)ws; ws += (size_t)EW0_E * 2;
    unsigned short* ew1p = (unsigned short*)ws; ws += (size_t)EW1_E * 2;
    unsigned short* ew2p = (unsigned short*)ws; ws += (size_t)EW1_E * 2;
    unsigned short* nw0p = (unsigned short*)ws; ws += (size_t)NW0_E * 2;
    unsigned short* nw1p = (unsigned short*)ws; ws += (size_t)NW1_E * 2;
    unsigned short* nw2p = (unsigned short*)ws; ws += (size_t)NW1_E * 2;

    // ---- init state ----
    hipMemcpyAsync(e_buf, edge_f, E_B, hipMemcpyDeviceToDevice, stream);
    hipMemcpyAsync(n_buf, node_f, N_B, hipMemcpyDeviceToDevice, stream);

    // ---- convert + fragment-pack all weights to bf16 (once per launch) ----
    pack_weights_kernel<<<(EW0_E + 255) / 256, 256, 0, stream>>>(ew0, ew0p, 3 * GP_D, EW0_E);
    pack_weights_kernel<<<(EW1_E + 255) / 256, 256, 0, stream>>>(ew1, ew1p, GP_D, EW1_E);
    pack_weights_kernel<<<(EW1_E + 255) / 256, 256, 0, stream>>>(ew2, ew2p, GP_D, EW1_E);
    pack_weights_kernel<<<(NW0_E + 255) / 256, 256, 0, stream>>>(nw0, nw0p, 2 * GP_D, NW0_E);
    pack_weights_kernel<<<(NW1_E + 255) / 256, 256, 0, stream>>>(nw1, nw1p, GP_D, NW1_E);
    pack_weights_kernel<<<(NW1_E + 255) / 256, 256, 0, stream>>>(nw2, nw2p, GP_D, NW1_E);

    const int edge_blocks = GP_EDGES / EBM;   // 2500
    const int node_blocks = GP_NODES / NBM;   // 625

    for (int i = 0; i < GP_P; ++i) {
        hipMemsetAsync(agg, 0, N_B, stream);
        edge_block_kernel<<<edge_blocks, 256, 0, stream>>>(
            n_buf, e_buf, agg, src, dst,
            ew0p + (size_t)i * 3 * GP_D * GP_D,
            ew1p + (size_t)i * GP_D * GP_D,
            ew2p + (size_t)i * GP_D * GP_D,
            eb0 + i * GP_D, eb1 + i * GP_D, eb2 + i * GP_D);
        node_block_kernel<<<node_blocks, 256, 0, stream>>>(
            n_buf, agg,
            nw0p + (size_t)i * 2 * GP_D * GP_D,
            nw1p + (size_t)i * GP_D * GP_D,
            nw2p + (size_t)i * GP_D * GP_D,
            nb0 + i * GP_D, nb1 + i * GP_D, nb2 + i * GP_D);
    }

    hipMemcpyAsync(d_out, n_buf, N_B, hipMemcpyDeviceToDevice, stream);
}
